// PerceiverEncoder_53712861004078
// MI455X (gfx1250) — matrix operand order repaired and checked by reading, not compiled, not run on hardware
//
#include <hip/hip_runtime.h>
#include <hip/hip_bf16.h>
#include <math.h>

// ---------------------------------------------------------------------------
// Perceiver encoder forward for MI455X (gfx1250, wave32, WMMA).
// All GEMMs run through v_wmma_f32_16x16x32_bf16 (fp32 -> bf16 at LDS staging,
// fp32 accumulate). GEMM is double-buffered / software-pipelined so global
// loads for tile t+1 overlap the WMMAs on tile t. Attention is a fused
// flash-style kernel (no NxM score tensor in HBM).
// ---------------------------------------------------------------------------

#define BB 4
#define MM 2048
#define CIN 768
#define NN 512
#define DD 1024
#define HH 8
#define HD 128
#define LL 6
#define FF 4096   // WF * D

typedef __attribute__((ext_vector_type(16))) __bf16 bf16x16;
typedef __attribute__((ext_vector_type(8)))  float  floatx8;

union Frag {
  bf16x16 h;
  uint4 q[2];
  unsigned short s[16];
};

__device__ __forceinline__ unsigned short cvt_bf16(float x) {
  union { float f; unsigned u; } v; v.f = x;
  unsigned r = v.u + 0x7FFFu + ((v.u >> 16) & 1u);   // round-to-nearest-even
  return (unsigned short)(r >> 16);
}

__device__ __forceinline__ floatx8 wmma_bf16(const Frag& a, const Frag& b, floatx8 c) {
  // 8 args: (neg_a, A, neg_b, B, c_mod, C, reuse_a, reuse_b)
  return __builtin_amdgcn_wmma_f32_16x16x32_bf16(false, a.h, false, b.h,
                                                 (short)0, c, false, false);
}

// ---------------------------------------------------------------------------
// Broadcast latent (N,D) -> (B,N,D)
// ---------------------------------------------------------------------------
__global__ void broadcast_kernel(const float* __restrict__ src,
                                 float* __restrict__ dst, int total, int per) {
  int i = blockIdx.x * 256 + threadIdx.x;
  if (i < total) dst[i] = src[i % per];
}

// ---------------------------------------------------------------------------
// LayerNorm over last dim C, one row per block (256 threads).
// ---------------------------------------------------------------------------
__global__ __launch_bounds__(256)
void layernorm_kernel(const float* __restrict__ x, const float* __restrict__ s,
                      const float* __restrict__ b, float* __restrict__ y, int C) {
  __shared__ float red[256];
  const int tid = threadIdx.x;
  const size_t row = blockIdx.x;
  const float* xr = x + row * (size_t)C;

  float sum = 0.f;
  for (int c = tid; c < C; c += 256) sum += xr[c];
  red[tid] = sum; __syncthreads();
  for (int st = 128; st > 0; st >>= 1) {
    if (tid < st) red[tid] += red[tid + st];
    __syncthreads();
  }
  const float mean = red[0] / (float)C;
  __syncthreads();

  float vs = 0.f;
  for (int c = tid; c < C; c += 256) { float d = xr[c] - mean; vs += d * d; }
  red[tid] = vs; __syncthreads();
  for (int st = 128; st > 0; st >>= 1) {
    if (tid < st) red[tid] += red[tid + st];
    __syncthreads();
  }
  const float rstd = rsqrtf(red[0] / (float)C + 1e-5f);

  float* yr = y + row * (size_t)C;
  for (int c = tid; c < C; c += 256)
    yr[c] = (xr[c] - mean) * rstd * s[c] + b[c];
}

// ---------------------------------------------------------------------------
// GEMM: out[M,N] = act( A[M,K] @ W[K,N] + bias ) (+ res)
// fp32 in/out, bf16 WMMA inside. Block 128x128 (8 waves of 64x32), K step 32,
// double-buffered LDS, software-pipelined global loads.
// Requires M%128==0, N%128==0, K%32==0 (true for all call sites).
// ---------------------------------------------------------------------------
#define G_BM 128
#define G_BN 128
#define G_BK 32
#define G_PA 40   // LDS pitch (ushorts) for A tile rows   (32 + pad, 16B mult)
#define G_PB 40   // LDS pitch (ushorts) for W^T tile rows

__device__ __forceinline__ void g_stage_load(const float* __restrict__ A,
                                             const float* __restrict__ W,
                                             int tid, int blockRow, int blockCol,
                                             int K, int Ncols, int k0,
                                             float4* ra, float4* rb) {
  for (int i = 0; i < 4; ++i) {
    int idx = tid + i * 256;                       // 0..1023 float4s
    int r = idx >> 3, c = (idx & 7) << 2;          // A: 128 rows x 8 float4
    ra[i] = *(const float4*)(A + (size_t)(blockRow + r) * K + k0 + c);
    int kk = idx >> 5, n = (idx & 31) << 2;        // W: 32 rows x 32 float4
    rb[i] = *(const float4*)(W + (size_t)(k0 + kk) * Ncols + blockCol + n);
  }
}

__device__ __forceinline__ void g_stage_store(unsigned short* __restrict__ dA,
                                              unsigned short* __restrict__ dB,
                                              int tid, const float4* ra,
                                              const float4* rb) {
  for (int i = 0; i < 4; ++i) {
    int idx = tid + i * 256;
    int r = idx >> 3, c = (idx & 7) << 2;
    unsigned short* d = &dA[r * G_PA + c];
    d[0] = cvt_bf16(ra[i].x); d[1] = cvt_bf16(ra[i].y);
    d[2] = cvt_bf16(ra[i].z); d[3] = cvt_bf16(ra[i].w);
    int kk = idx >> 5, n = (idx & 31) << 2;
    dB[(n + 0) * G_PB + kk] = cvt_bf16(rb[i].x);
    dB[(n + 1) * G_PB + kk] = cvt_bf16(rb[i].y);
    dB[(n + 2) * G_PB + kk] = cvt_bf16(rb[i].z);
    dB[(n + 3) * G_PB + kk] = cvt_bf16(rb[i].w);
  }
}

__global__ __launch_bounds__(256)
void gemm_kernel(const float* __restrict__ A, const float* __restrict__ W,
                 const float* __restrict__ bias, const float* __restrict__ res,
                 float* __restrict__ out, int Mrows, int K, int Ncols,
                 int fuse_gelu) {
  __shared__ __align__(16) unsigned short sA[2][G_BM * G_PA];
  __shared__ __align__(16) unsigned short sB[2][G_BN * G_PB];  // W^T : [n][k]

  const int tid  = threadIdx.x;
  const int wave = tid >> 5, lane = tid & 31;
  const int lrow = lane & 15, lhalf = lane >> 4;
  const int waveM = wave >> 2;          // 0..1  (64 rows each)
  const int waveN = wave & 3;           // 0..3  (32 cols each)
  const int blockRow = blockIdx.y * G_BM;
  const int blockCol = blockIdx.x * G_BN;

  floatx8 acc[4][2];
  for (int i = 0; i < 4; ++i)
    for (int j = 0; j < 2; ++j)
      for (int r = 0; r < 8; ++r) acc[i][j][r] = 0.f;

  // prologue: stage tile 0 into buffer 0
  {
    float4 ra[4], rb[4];
    g_stage_load(A, W, tid, blockRow, blockCol, K, Ncols, 0, ra, rb);
    g_stage_store(sA[0], sB[0], tid, ra, rb);
  }
  __syncthreads();

  const int nk = K / G_BK;
  for (int t = 0; t < nk; ++t) {
    const int cur = t & 1, nxt = cur ^ 1;
    const bool have_next = (t + 1 < nk);

    // issue global loads for tile t+1 first (overlap with WMMA below)
    float4 ra[4], rb[4];
    if (have_next)
      g_stage_load(A, W, tid, blockRow, blockCol, K, Ncols, (t + 1) * G_BK,
                   ra, rb);
    if (t + 2 < nk) {  // speculative prefetch of tile t+2 -> global_prefetch_b8
      __builtin_prefetch(&A[(size_t)(blockRow + (tid & 127)) * K + (t + 2) * G_BK], 0, 1);
      __builtin_prefetch(&W[(size_t)((t + 2) * G_BK + (tid & 31)) * Ncols + blockCol +
                            ((tid >> 5) << 4)], 0, 1);
    }

    // fragments + WMMA on tile t (buffer cur)
    Frag af[4], bfr[2];
    for (int mi = 0; mi < 4; ++mi) {
      int row = waveM * 64 + mi * 16 + lrow;
      af[mi].q[0] = *(const uint4*)&sA[cur][row * G_PA + lhalf * 8];
      af[mi].q[1] = *(const uint4*)&sA[cur][row * G_PA + 16 + lhalf * 8];
    }
    for (int ni = 0; ni < 2; ++ni) {
      int col = waveN * 32 + ni * 16 + lrow;
      bfr[ni].q[0] = *(const uint4*)&sB[cur][col * G_PB + lhalf * 8];
      bfr[ni].q[1] = *(const uint4*)&sB[cur][col * G_PB + 16 + lhalf * 8];
    }
    for (int mi = 0; mi < 4; ++mi)
      for (int ni = 0; ni < 2; ++ni)
        acc[mi][ni] = wmma_bf16(af[mi], bfr[ni], acc[mi][ni]);

    // convert + store tile t+1 into the other buffer
    if (have_next)
      g_stage_store(sA[nxt], sB[nxt], tid, ra, rb);
    __syncthreads();
  }

  // ---- epilogue: bias (+gelu) (+residual) ---------------------------------
  for (int ni = 0; ni < 2; ++ni) {
    int col = blockCol + waveN * 32 + ni * 16 + lrow;
    float bv = bias ? bias[col] : 0.f;
    for (int mi = 0; mi < 4; ++mi) {
      int rbase = blockRow + waveM * 64 + mi * 16 + 8 * lhalf;
      for (int r = 0; r < 8; ++r) {
        float v = acc[mi][ni][r] + bv;
        if (fuse_gelu) v = 0.5f * v * (1.f + erff(v * 0.70710678118654752f));
        size_t o = (size_t)(rbase + r) * Ncols + col;
        if (res) v += res[o];
        out[o] = v;
      }
    }
  }
}

// ---------------------------------------------------------------------------
// Fused flash attention (head dim 128, D=1024, H=8).
// q: (B,Nq,D), k/v: (B,Mk,D), out: (B,Nq,D) -- all fp32.
// grid = (Nq/64, B*H), block = 128 (4 waves, 16 query rows per wave).
// mask: int32 per (b, key); nonzero = padded (cross-attn only, else null).
// ---------------------------------------------------------------------------
#define A_PK 136   // LDS pitch ushorts for K tile rows   [key][d]   (128+8)
#define A_PV 72    // LDS pitch ushorts for V^T tile rows [d][key]   (64+8)
#define A_PS 68    // LDS pitch floats  for S rows        (64+4)
#define A_PP 72    // LDS pitch ushorts for P rows        (64+8)

__global__ __launch_bounds__(128)
void attn_kernel(const float* __restrict__ q, const float* __restrict__ k,
                 const float* __restrict__ v, float* __restrict__ out,
                 const int* __restrict__ mask, int Nq, int Mk) {
  __shared__ __align__(16) unsigned short sK[64 * A_PK];
  __shared__ __align__(16) unsigned short sV[HD * A_PV];
  __shared__ __align__(16) float          sS[4 * 16 * A_PS];
  __shared__ __align__(16) unsigned short sP[4 * 16 * A_PP];
  __shared__ float sAl[4 * 16];

  const int tid  = threadIdx.x;
  const int wave = tid >> 5, lane = tid & 31;
  const int lrow = lane & 15, lhalf = lane >> 4;
  const int b = blockIdx.y >> 3;     // H == 8
  const int h = blockIdx.y & 7;

  const float* qb = q + (size_t)b * Nq * DD + (size_t)h * HD;
  const float* kb = k + (size_t)b * Mk * DD + (size_t)h * HD;
  const float* vb = v + (size_t)b * Mk * DD + (size_t)h * HD;
  const int* mb = mask ? (mask + (size_t)b * Mk) : nullptr;

  // ---- preload Q fragments (16 rows x 128, scale folded in) ---------------
  const float scale = 0.08838834764831845f;   // 128^-0.5
  const int qrow = blockIdx.x * 64 + wave * 16 + lrow;
  const float* qp = qb + (size_t)qrow * DD;
  Frag qf[4];
  for (int s = 0; s < 4; ++s) {
    int d0 = s * 32 + lhalf * 8;
    int d1 = s * 32 + 16 + lhalf * 8;
    for (int i = 0; i < 8; ++i) qf[s].s[i]     = cvt_bf16(qp[d0 + i] * scale);
    for (int i = 0; i < 8; ++i) qf[s].s[8 + i] = cvt_bf16(qp[d1 + i] * scale);
  }

  floatx8 oacc[8];
  for (int n = 0; n < 8; ++n)
    for (int r = 0; r < 8; ++r) oacc[n][r] = 0.f;
  float m_run = -3.0e38f, l_run = 0.f;   // meaningful in lanes 0..15

  const int nkt = Mk >> 6;
  for (int kc = 0; kc < nkt; ++kc) {
    // ---- stage K tile [key][d] and V tile transposed [d][key] -------------
    // batch 8 b128 loads in flight before converting (hide HBM latency)
    for (int i = 0; i < 4; ++i) {
      float4 rk[4], rv[4];
      int rr[4], ff[4];
      for (int j = 0; j < 4; ++j) {
        int idx = tid + (i * 4 + j) * 128;         // 0..2047 float4s
        rr[j] = idx >> 5; ff[j] = (idx & 31) << 2;
        size_t g = (size_t)(kc * 64 + rr[j]) * DD + ff[j];
        rk[j] = *(const float4*)(kb + g);
        rv[j] = *(const float4*)(vb + g);
      }
      for (int j = 0; j < 4; ++j) {
        unsigned short* d = &sK[rr[j] * A_PK + ff[j]];
        d[0] = cvt_bf16(rk[j].x); d[1] = cvt_bf16(rk[j].y);
        d[2] = cvt_bf16(rk[j].z); d[3] = cvt_bf16(rk[j].w);
        sV[(ff[j] + 0) * A_PV + rr[j]] = cvt_bf16(rv[j].x);
        sV[(ff[j] + 1) * A_PV + rr[j]] = cvt_bf16(rv[j].y);
        sV[(ff[j] + 2) * A_PV + rr[j]] = cvt_bf16(rv[j].z);
        sV[(ff[j] + 3) * A_PV + rr[j]] = cvt_bf16(rv[j].w);
      }
    }
    __syncthreads();

    // ---- S = Q K^T (16 x 64 per wave) -------------------------------------
    floatx8 sacc[4];
    for (int n = 0; n < 4; ++n)
      for (int r = 0; r < 8; ++r) sacc[n][r] = 0.f;
    for (int s = 0; s < 4; ++s) {
      for (int n = 0; n < 4; ++n) {
        Frag kf;
        const unsigned short* kp = &sK[(n * 16 + lrow) * A_PK + s * 32 + lhalf * 8];
        kf.q[0] = *(const uint4*)kp;
        kf.q[1] = *(const uint4*)(kp + 16);
        sacc[n] = wmma_bf16(qf[s], kf, sacc[n]);
      }
    }
    // ---- key padding mask --------------------------------------------------
    if (mb) {
      for (int n = 0; n < 4; ++n) {
        int key = kc * 64 + n * 16 + lrow;
        if (mb[key] != 0)
          for (int r = 0; r < 8; ++r) sacc[n][r] = -3.0e38f;
      }
    }
    // ---- spill S (D-layout) to LDS -----------------------------------------
    for (int n = 0; n < 4; ++n)
      for (int r = 0; r < 8; ++r)
        sS[(wave * 16 + r + 8 * lhalf) * A_PS + n * 16 + lrow] = sacc[n][r];
    __syncthreads();

    // ---- online softmax (one row per lane 0..15 of each wave) -------------
    if (lane < 16) {
      const float* srow = &sS[(wave * 16 + lane) * A_PS];
      float mx = -3.0e38f;
      for (int j = 0; j < 64; ++j) mx = fmaxf(mx, srow[j]);
      float mnew = fmaxf(m_run, mx);
      float alpha = __expf(m_run - mnew);
      unsigned short* prow = &sP[(wave * 16 + lane) * A_PP];
      float psum = 0.f;
      for (int j = 0; j < 64; ++j) {
        float p = __expf(srow[j] - mnew);
        psum += p;
        prow[j] = cvt_bf16(p);
      }
      l_run = l_run * alpha + psum;
      m_run = mnew;
      sAl[wave * 16 + lane] = alpha;
    }
    __syncthreads();

    // ---- rescale O, then O += P V -----------------------------------------
    float alr[8];
    for (int r = 0; r < 8; ++r) alr[r] = sAl[wave * 16 + 8 * lhalf + r];
    for (int n = 0; n < 8; ++n)
      for (int r = 0; r < 8; ++r) oacc[n][r] *= alr[r];

    for (int ks = 0; ks < 2; ++ks) {
      Frag pf;
      const unsigned short* pp = &sP[(wave * 16 + lrow) * A_PP + ks * 32];
      pf.q[0] = *(const uint4*)(pp + lhalf * 8);
      pf.q[1] = *(const uint4*)(pp + 16 + lhalf * 8);
      for (int n = 0; n < 8; ++n) {
        Frag vf;
        const unsigned short* vp = &sV[(n * 16 + lrow) * A_PV + ks * 32 + lhalf * 8];
        vf.q[0] = *(const uint4*)vp;
        vf.q[1] = *(const uint4*)(vp + 16);
        oacc[n] = wmma_bf16(pf, vf, oacc[n]);
      }
    }
    __syncthreads();
  }

  // ---- normalize and store ------------------------------------------------
  if (lane < 16) sAl[wave * 16 + lane] = (l_run > 0.f) ? (1.f / l_run) : 0.f;
  __syncthreads();
  float linv[8];
  for (int r = 0; r < 8; ++r) linv[r] = sAl[wave * 16 + 8 * lhalf + r];

  float* ob = out + (size_t)b * Nq * DD + (size_t)h * HD;
  for (int n = 0; n < 8; ++n)
    for (int r = 0; r < 8; ++r) {
      int row = blockIdx.x * 64 + wave * 16 + 8 * lhalf + r;
      ob[(size_t)row * DD + n * 16 + lrow] = oacc[n][r] * linv[r];
    }
}

// ---------------------------------------------------------------------------
// Host-side orchestration
// ---------------------------------------------------------------------------
static inline void run_gemm(const float* A, const float* W, const float* bias,
                            const float* res, float* out, int M, int K, int N,
                            int gelu, hipStream_t st) {
  dim3 grid(N / G_BN, M / G_BM);
  gemm_kernel<<<grid, 256, 0, st>>>(A, W, bias, res, out, M, K, N, gelu);
}

static inline void run_ln(const float* x, const float* s, const float* b,
                          float* y, int rows, int C, hipStream_t st) {
  layernorm_kernel<<<rows, 256, 0, st>>>(x, s, b, y, C);
}

static inline void run_attn(const float* q, const float* k, const float* v,
                            float* o, const int* mask, int Nq, int Mk,
                            hipStream_t st) {
  dim3 grid(Nq / 64, BB * HH);
  attn_kernel<<<grid, 128, 0, st>>>(q, k, v, o, mask, Nq, Mk);
}

extern "C" void kernel_launch(void* const* d_in, const int* in_sizes, int n_in,
                              void* d_out, int out_size, void* d_ws, size_t ws_size,
                              hipStream_t stream) {
  (void)in_sizes; (void)n_in; (void)ws_size;
  int i = 0;
  const float* x        = (const float*)d_in[i++];   // (B,M,CIN)
  const int*   pad      = (const int*)  d_in[i++];   // (B,M) bool->int
  const float* latent   = (const float*)d_in[i++];   // (N,D)
  const float* ca_q_ln_s  = (const float*)d_in[i++];
  const float* ca_q_ln_b  = (const float*)d_in[i++];
  const float* ca_kv_ln_s = (const float*)d_in[i++];
  const float* ca_kv_ln_b = (const float*)d_in[i++];
  const float* ca_wq = (const float*)d_in[i++]; const float* ca_bq = (const float*)d_in[i++];
  const float* ca_wk = (const float*)d_in[i++]; const float* ca_bk = (const float*)d_in[i++];
  const float* ca_wv = (const float*)d_in[i++]; const float* ca_bv = (const float*)d_in[i++];
  const float* ca_wo = (const float*)d_in[i++]; const float* ca_bo = (const float*)d_in[i++];
  const float* ca_mlp_ln_s = (const float*)d_in[i++];
  const float* ca_mlp_ln_b = (const float*)d_in[i++];
  const float* ca_w1 = (const float*)d_in[i++]; const float* ca_b1 = (const float*)d_in[i++];
  const float* ca_w2 = (const float*)d_in[i++]; const float* ca_b2 = (const float*)d_in[i++];
  const float* sa_ln_s = (const float*)d_in[i++]; const float* sa_ln_b = (const float*)d_in[i++];
  const float* sa_wq = (const float*)d_in[i++]; const float* sa_bq = (const float*)d_in[i++];
  const float* sa_wk = (const float*)d_in[i++]; const float* sa_bk = (const float*)d_in[i++];
  const float* sa_wv = (const float*)d_in[i++]; const float* sa_bv = (const float*)d_in[i++];
  const float* sa_wo = (const float*)d_in[i++]; const float* sa_bo = (const float*)d_in[i++];
  const float* sa_mlp_ln_s = (const float*)d_in[i++];
  const float* sa_mlp_ln_b = (const float*)d_in[i++];
  const float* sa_w1 = (const float*)d_in[i++]; const float* sa_b1 = (const float*)d_in[i++];
  const float* sa_w2 = (const float*)d_in[i++]; const float* sa_b2 = (const float*)d_in[i++];

  // workspace layout (floats); total ~151 MB
  float* ws   = (float*)d_ws;
  float* lat  = ws;                                   // B*N*D     = 2,097,152
  float* nrm  = lat + (size_t)BB * NN * DD;           // B*M*CIN   = 6,291,456
  float* qb   = nrm + (size_t)BB * MM * CIN;          // B*N*D
  float* kb   = qb  + (size_t)BB * NN * DD;           // B*M*D     = 8,388,608
  float* vbuf = kb  + (size_t)BB * MM * DD;           // B*M*D
  float* ob   = vbuf + (size_t)BB * MM * DD;          // B*N*D
  float* hb   = ob  + (size_t)BB * NN * DD;           // B*N*FF    = 8,388,608

  const int latElems = BB * NN * DD;

  // latents broadcast
  broadcast_kernel<<<(latElems + 255) / 256, 256, 0, stream>>>(latent, lat,
                                                               latElems, NN * DD);
  // ---- cross attention ----------------------------------------------------
  run_ln(lat, ca_q_ln_s, ca_q_ln_b, ob, BB * NN, DD, stream);      // q_in -> ob
  run_ln(x,   ca_kv_ln_s, ca_kv_ln_b, nrm, BB * MM, CIN, stream);  // kv_in -> nrm
  run_gemm(ob,  ca_wq, ca_bq, nullptr, qb,   BB * NN, DD,  DD, 0, stream);
  run_gemm(nrm, ca_wk, ca_bk, nullptr, kb,   BB * MM, CIN, DD, 0, stream);
  run_gemm(nrm, ca_wv, ca_bv, nullptr, vbuf, BB * MM, CIN, DD, 0, stream);
  run_attn(qb, kb, vbuf, ob, pad, NN, MM, stream);
  run_gemm(ob, ca_wo, ca_bo, lat, lat, BB * NN, DD, DD, 0, stream);
  // cross MLP
  run_ln(lat, ca_mlp_ln_s, ca_mlp_ln_b, nrm, BB * NN, DD, stream);
  run_gemm(nrm, ca_w1, ca_b1, nullptr, hb, BB * NN, DD, FF, 1, stream);
  run_gemm(hb,  ca_w2, ca_b2, lat, lat,  BB * NN, FF, DD, 0, stream);

  // ---- self attention layers ---------------------------------------------
  for (int l = 0; l < LL; ++l) {
    const size_t wO = (size_t)l * DD * DD;
    const size_t bO = (size_t)l * DD;
    run_ln(lat, sa_ln_s + bO, sa_ln_b + bO, nrm, BB * NN, DD, stream);
    run_gemm(nrm, sa_wq + wO, sa_bq + bO, nullptr, qb,   BB * NN, DD, DD, 0, stream);
    run_gemm(nrm, sa_wk + wO, sa_bk + bO, nullptr, kb,   BB * NN, DD, DD, 0, stream);
    run_gemm(nrm, sa_wv + wO, sa_bv + bO, nullptr, vbuf, BB * NN, DD, DD, 0, stream);
    run_attn(qb, kb, vbuf, ob, nullptr, NN, NN, stream);
    run_gemm(ob, sa_wo + wO, sa_bo + bO, lat, lat, BB * NN, DD, DD, 0, stream);
    run_ln(lat, sa_mlp_ln_s + bO, sa_mlp_ln_b + bO, nrm, BB * NN, DD, stream);
    run_gemm(nrm, sa_w1 + (size_t)l * DD * FF, sa_b1 + (size_t)l * FF,
             nullptr, hb, BB * NN, DD, FF, 1, stream);
    run_gemm(hb, sa_w2 + (size_t)l * FF * DD, sa_b2 + bO,
             lat, lat, BB * NN, FF, DD, 0, stream);
  }

  hipMemcpyAsync(d_out, lat, (size_t)out_size * sizeof(float),
                 hipMemcpyDeviceToDevice, stream);
}
